// FastFeedForward_4827543241143
// MI455X (gfx1250) — matrix operand order repaired and checked by reading, not compiled, not run on hardware
//
#include <hip/hip_runtime.h>
#include <math.h>

#define DIM   512
#define HID   2048
#define NEXP  8
#define NTOK  4096
#define BM    32
#define CHUNK 128

typedef __attribute__((ext_vector_type(16))) __bf16        v16bf;
typedef __attribute__((ext_vector_type(8)))  float         v8f;
typedef __attribute__((ext_vector_type(4)))  unsigned int  v4u;

union BF16Vec { v16bf v; v4u u[2]; };

__device__ __forceinline__ unsigned short f2bf(float f) {
  union { float f; unsigned int u; } x; x.f = f;
  unsigned int u = x.u;
  u += 0x7FFFu + ((u >> 16) & 1u);   // round-to-nearest-even
  return (unsigned short)(u >> 16);
}

__device__ __forceinline__ float gelu_erf(float x) {
  return 0.5f * x * (1.0f + erff(x * 0.70710678118654752f));
}

// -------------------------------------------------------------------------
// Pack f32 weights [E, K, N] into bf16 WMMA-B tiles.
// Tile = 32(K) x 16(N). Per tile, lane L owns column (L&15), K-half (L>>4):
// 16 consecutive-K bf16 values stored contiguously at tile*512 + L*16.
// This makes the WMMA B operand two coalesced global_load_b128 per lane.
// -------------------------------------------------------------------------
__global__ __launch_bounds__(256) void pack_bf16_tiles(
    const float* __restrict__ src, unsigned short* __restrict__ dst,
    int K, int N) {
  int idx = blockIdx.x * 256 + threadIdx.x;
  int total = NEXP * (K >> 5) * (N >> 4) * 32;
  if (idx >= total) return;
  int L = idx & 31;
  int t = idx >> 5;
  int ntiles = N >> 4;
  int nt = t % ntiles;
  int rest = t / ntiles;
  int ktiles = K >> 5;
  int kt = rest % ktiles;
  int e  = rest / ktiles;
  int khalf = L >> 4, col = L & 15;
  const float* s = src + ((size_t)e * K + (size_t)kt * 32 + khalf * 16) * N
                       + (size_t)nt * 16 + col;
  unsigned int w[8];
#pragma unroll
  for (int j = 0; j < 8; ++j) {
    unsigned int lo = f2bf(s[(size_t)(2 * j) * N]);
    unsigned int hiw = f2bf(s[(size_t)(2 * j + 1) * N]);
    w[j] = lo | (hiw << 16);
  }
  v4u* d = (v4u*)(dst + (size_t)t * 512 + L * 16);
  v4u a = {w[0], w[1], w[2], w[3]};
  v4u b = {w[4], w[5], w[6], w[7]};
  d[0] = a;
  d[1] = b;
}

// -------------------------------------------------------------------------
// Router: one wave per token. 7 dot products, shfl reduce, binary-tree
// leaf probabilities (exact sigmoid).
// -------------------------------------------------------------------------
__global__ __launch_bounds__(256) void router_kernel(
    const float* __restrict__ x, const float* __restrict__ rw,
    const float* __restrict__ rb, float* __restrict__ leaf) {
  int wave = threadIdx.x >> 5, lane = threadIdx.x & 31;
  int tok = blockIdx.x * 8 + wave;
  float acc[7];
#pragma unroll
  for (int r = 0; r < 7; ++r) acc[r] = 0.0f;
  const float* xr = x + (size_t)tok * DIM;
  for (int k = lane; k < DIM; k += 32) {
    float xv = xr[k];
#pragma unroll
    for (int r = 0; r < 7; ++r) acc[r] += xv * rw[r * DIM + k];
  }
#pragma unroll
  for (int r = 0; r < 7; ++r) {
#pragma unroll
    for (int off = 16; off > 0; off >>= 1)
      acc[r] += __shfl_xor(acc[r], off, 32);
  }
  if (lane == 0) {
    float p[7];
#pragma unroll
    for (int r = 0; r < 7; ++r)
      p[r] = 1.0f / (1.0f + expf(-(acc[r] + rb[r])));
#pragma unroll
    for (int l = 0; l < 8; ++l) {
      int a = (l >> 2) & 1, b = (l >> 1) & 1, c = l & 1;
      float pa = a ? p[0] : 1.0f - p[0];
      float pb = b ? p[1 + a] : 1.0f - p[1 + a];
      float pc = c ? p[3 + 2 * a + b] : 1.0f - p[3 + 2 * a + b];
      leaf[(size_t)tok * NEXP + l] = pa * pb * pc;
    }
  }
}

// -------------------------------------------------------------------------
// Fused soft-MoE FFN. Block = 32 tokens, 8 waves, all experts.
//   Phase A: h_chunk[32x128] = gelu(x@w1 + b1) * leaf_prob   (into LDS bf16)
//   Phase B: out_acc[32x512] += h_chunk @ w2                 (regs, f32)
// out = out_acc + sum_e p_e * b2[e]
// -------------------------------------------------------------------------
__global__ __launch_bounds__(256) void ffn_moe_kernel(
    const float* __restrict__ x,
    const unsigned short* __restrict__ w1p,
    const float* __restrict__ b1,
    const unsigned short* __restrict__ w2p,
    const float* __restrict__ b2,
    const float* __restrict__ leaf,
    float* __restrict__ out) {
  __shared__ unsigned short xs[BM][DIM + 8];    // byte stride 1040: 16B-aligned, bank-staggered
  __shared__ unsigned short hs[BM][CHUNK + 8];  // byte stride  272: 16B-aligned, bank-staggered
  __shared__ float ps[BM][NEXP];

  const int tid  = threadIdx.x;
  const int wave = tid >> 5, lane = tid & 31;
  const int tokBase = blockIdx.x * BM;
  const int nl = lane & 15;   // lane's column within a 16-wide tile
  const int hi = lane >> 4;   // K/M half select per WMMA lane layout

  // ---- stage x tile as bf16 in LDS ----
  for (int i = tid; i < BM * (DIM / 4); i += 256) {
    int row = i >> 7;                  // DIM/4 = 128 float4 per row
    int c4 = (i & 127) * 4;
    float4 v = *(const float4*)(x + (size_t)(tokBase + row) * DIM + c4);
    xs[row][c4 + 0] = f2bf(v.x);
    xs[row][c4 + 1] = f2bf(v.y);
    xs[row][c4 + 2] = f2bf(v.z);
    xs[row][c4 + 3] = f2bf(v.w);
  }
  if (tid < BM * NEXP) {
    int row = tid >> 3;
    ps[row][tid & 7] = leaf[(size_t)(tokBase + row) * NEXP + (tid & 7)];
  }
  __syncthreads();

  // Phase-B wave tile: rows 16*mi, cols 128*di of the 32x512 output.
  const int mi = wave & 1, di = wave >> 1;
  // Phase-A wave tile: rows 16*ami, cols 32*ani of the 32x128 h chunk.
  const int ami = wave & 1, ani = wave >> 1;

  v8f zero;
#pragma unroll
  for (int r = 0; r < 8; ++r) zero[r] = 0.0f;
  v8f acc[8];
#pragma unroll
  for (int j = 0; j < 8; ++j) acc[j] = zero;

  for (int e = 0; e < NEXP; ++e) {
    for (int c = 0; c < HID / CHUNK; ++c) {
      const int hk0 = c * CHUNK;
      __syncthreads();   // hs consumed by previous Phase B, safe to overwrite

      // ---------------- Phase A ----------------
      const int col0 = hk0 + 32 * ani;
      const float bias0 = b1[e * HID + col0 + nl];
      const float bias1 = b1[e * HID + col0 + 16 + nl];
      v8f h0, h1;
#pragma unroll
      for (int r = 0; r < 8; ++r) { h0[r] = bias0; h1[r] = bias1; }

      const int am = 16 * ami + nl;
#pragma unroll 4
      for (int k0 = 0; k0 < DIM; k0 += 32) {
        BF16Vec A;
        const unsigned short* ap = &xs[am][k0 + 16 * hi];
        A.u[0] = *(const v4u*)ap;
        A.u[1] = *(const v4u*)(ap + 8);
        const size_t tbase =
            (((size_t)e * (DIM / 32) + (k0 >> 5)) * (HID / 16) + (col0 >> 4)) * 512
            + (size_t)lane * 16;
        BF16Vec B0, B1;
        const v4u* bp0 = (const v4u*)(w1p + tbase);
        B0.u[0] = bp0[0]; B0.u[1] = bp0[1];
        const v4u* bp1 = (const v4u*)(w1p + tbase + 512);
        B1.u[0] = bp1[0]; B1.u[1] = bp1[1];
        h0 = __builtin_amdgcn_wmma_f32_16x16x32_bf16(false, A.v, false, B0.v,
                                                     (short)0, h0, false, false);
        h1 = __builtin_amdgcn_wmma_f32_16x16x32_bf16(false, A.v, false, B1.v,
                                                     (short)0, h1, false, false);
      }
      // gelu, scale by this expert's leaf prob (linear in h rows), to LDS bf16
#pragma unroll
      for (int r = 0; r < 8; ++r) {
        const int tr = 16 * ami + 8 * hi + r;
        const float pscale = ps[tr][e];
        hs[tr][32 * ani + nl]      = f2bf(gelu_erf(h0[r]) * pscale);
        hs[tr][32 * ani + 16 + nl] = f2bf(gelu_erf(h1[r]) * pscale);
      }
      __syncthreads();

      // ---------------- Phase B ----------------
      const int bm = 16 * mi + nl;
#pragma unroll
      for (int kk = 0; kk < CHUNK; kk += 32) {
        BF16Vec A;
        const unsigned short* ap = &hs[bm][kk + 16 * hi];
        A.u[0] = *(const v4u*)ap;
        A.u[1] = *(const v4u*)(ap + 8);
        const size_t tb2 =
            (((size_t)e * (HID / 32) + ((hk0 + kk) >> 5)) * (DIM / 16) + 8 * di) * 512
            + (size_t)lane * 16;
#pragma unroll
        for (int j = 0; j < 8; ++j) {
          BF16Vec B;
          const v4u* bp = (const v4u*)(w2p + tb2 + (size_t)j * 512);
          B.u[0] = bp[0]; B.u[1] = bp[1];
          acc[j] = __builtin_amdgcn_wmma_f32_16x16x32_bf16(false, A.v, false, B.v,
                                                           (short)0, acc[j], false, false);
        }
      }
    }
  }

  // ---- epilogue: + sum_e p_e * b2[e], write out ----
#pragma unroll
  for (int j = 0; j < 8; ++j) {
    const int d = 128 * di + 16 * j + nl;
    float b2v[NEXP];
#pragma unroll
    for (int e = 0; e < NEXP; ++e) b2v[e] = b2[e * DIM + d];
#pragma unroll
    for (int r = 0; r < 8; ++r) {
      const int tr = 16 * mi + 8 * hi + r;
      float s = 0.0f;
#pragma unroll
      for (int e = 0; e < NEXP; ++e) s += ps[tr][e] * b2v[e];
      out[(size_t)(tokBase + tr) * DIM + d] = acc[j][r] + s;
    }
  }
}

extern "C" void kernel_launch(void* const* d_in, const int* in_sizes, int n_in,
                              void* d_out, int out_size, void* d_ws, size_t ws_size,
                              hipStream_t stream) {
  (void)in_sizes; (void)n_in; (void)out_size; (void)ws_size;
  const float* x  = (const float*)d_in[0];
  const float* rw = (const float*)d_in[1];
  const float* rb = (const float*)d_in[2];
  const float* w1 = (const float*)d_in[3];
  const float* b1 = (const float*)d_in[4];
  const float* w2 = (const float*)d_in[5];
  const float* b2 = (const float*)d_in[6];
  float* out = (float*)d_out;

  // Workspace layout: w1 bf16 tiles (16 MB), w2 bf16 tiles (16 MB), leaf (128 KB)
  unsigned short* w1p = (unsigned short*)d_ws;
  unsigned short* w2p = w1p + (size_t)NEXP * DIM * HID;
  float* leaf = (float*)(w2p + (size_t)NEXP * HID * DIM);

  const int lanes1 = NEXP * (DIM / 32) * (HID / 16) * 32;   // 524288
  const int lanes2 = NEXP * (HID / 32) * (DIM / 16) * 32;   // 524288
  hipLaunchKernelGGL(pack_bf16_tiles, dim3(lanes1 / 256), dim3(256), 0, stream,
                     w1, w1p, DIM, HID);
  hipLaunchKernelGGL(pack_bf16_tiles, dim3(lanes2 / 256), dim3(256), 0, stream,
                     w2, w2p, HID, DIM);
  hipLaunchKernelGGL(router_kernel, dim3(NTOK / 8), dim3(256), 0, stream,
                     x, rw, rb, leaf);
  hipLaunchKernelGGL(ffn_moe_kernel, dim3(NTOK / BM), dim3(256), 0, stream,
                     x, w1p, b1, w2p, b2, leaf, out);
}
